// unit_coupled_29240137351870
// MI455X (gfx1250) — matrix operand order repaired and checked by reading, not compiled, not run on hardware
//
#include <hip/hip_runtime.h>
#include <hip/hip_bf16.h>

#define NH 6
#define VJ 25
#define FF 64
#define TT 1600
#define DIM 216
#define HD 36
#define HDP 64
#define HDP3 48

typedef __attribute__((ext_vector_type(16))) _Float16 v16h;
typedef __attribute__((ext_vector_type(8)))  _Float16 v8h;
typedef __attribute__((ext_vector_type(8)))  float    v8f;

#if __has_builtin(__builtin_amdgcn_sched_barrier)
#define SCHED_SPLIT() __builtin_amdgcn_sched_barrier(0)
#else
#define SCHED_SPLIT() asm volatile("" ::: "memory")
#endif

// ---- WMMA helpers (wave32, f16 inputs, f32 accum) ----
__device__ __forceinline__ v8f wmma_f16(v16h a, v16h b, v8f c) {
    return __builtin_amdgcn_wmma_f32_16x16x32_f16(false, a, false, b, (short)0, c,
                                                  false, false);
}

// A-fragment: 16x32 f16 tile, row-major with leading dim ld (halfs).
// lane: m=lane&15, g=lane>>4 ; k(j) = {j + 8g} for j<8, {16 + (j-8) + 8g} for j>=8
__device__ __forceinline__ v16h load_afrag(const _Float16* base, int ld) {
    int lane = threadIdx.x & 31;
    int m = lane & 15, g = lane >> 4;
    const _Float16* p = base + (size_t)m * ld + 8 * g;
    v8h lo = *(const v8h*)p;
    v8h hi = *(const v8h*)(p + 16);
    v16h r;
#pragma unroll
    for (int j = 0; j < 8; ++j) { r[j] = lo[j]; r[j + 8] = hi[j]; }
    return r;
}

// B-fragment: 32x16 f16 tile, element(k,n) = base[n*ld + k].
// lane: n=lane&15, g=lane>>4 ; k(j) = {j + 8g} for j<8, {16 + (j-8) + 8g} for j>=8
__device__ __forceinline__ v16h load_bfrag(const _Float16* base, int ld) {
    int lane = threadIdx.x & 31;
    int n = lane & 15, g = lane >> 4;
    const _Float16* p = base + (size_t)n * ld + 8 * g;
    v8h lo = *(const v8h*)p;
    v8h hi = *(const v8h*)(p + 16);
    v16h r;
#pragma unroll
    for (int j = 0; j < 8; ++j) { r[j] = lo[j]; r[j + 8] = hi[j]; }
    return r;
}

// ---------------- K0: bias tables ----------------
__global__ void k_tables(const float* t_w1, const float* t_b1, const float* t_w2,
                         const float* t_b2, const float* h_w1, const float* h_b1,
                         const float* h_w2, const float* h_b2, const int* hop,
                         float* tbl_t, float* bias_h) {
    __shared__ float th[16][NH];
    int tid = threadIdx.x;
    if (tid < 2 * FF - 1) {
        float coord = (float)(tid - (FF - 1));
        float acc[NH] = {0, 0, 0, 0, 0, 0};
        for (int d = 0; d < DIM; ++d) {
            float r = fmaxf(coord * t_w1[d] + t_b1[d], 0.0f);
#pragma unroll
            for (int h = 0; h < NH; ++h) acc[h] += r * t_w2[h * DIM + d];
        }
        for (int h = 0; h < NH; ++h) tbl_t[tid * NH + h] = acc[h] + t_b2[h];
    }
    if (tid < 16) {
        float hv = (float)tid;
        float acc[NH] = {0, 0, 0, 0, 0, 0};
        for (int d = 0; d < DIM; ++d) {
            float r = fmaxf(hv * h_w1[d] + h_b1[d], 0.0f);
#pragma unroll
            for (int h = 0; h < NH; ++h) acc[h] += r * h_w2[h * DIM + d];
        }
        for (int h = 0; h < NH; ++h) th[tid][h] = acc[h] + h_b2[h];
    }
    __syncthreads();
    for (int idx = tid; idx < NH * VJ * VJ; idx += blockDim.x) {
        int h = idx / (VJ * VJ), ij = idx % (VJ * VJ);
        bias_h[idx] = th[hop[ij]][h];
    }
}

// ---------------- weight fp32 -> fp16 with zero padding ----------------
__global__ void k_convpad(const float* src, _Float16* dst, int R, int C, int Rp, int Cp) {
    size_t tot = (size_t)Rp * Cp;
    for (size_t idx = (size_t)blockIdx.x * blockDim.x + threadIdx.x; idx < tot;
         idx += (size_t)gridDim.x * blockDim.x) {
        int r = (int)(idx / Cp), c = (int)(idx % Cp);
        dst[idx] = (r < R && c < C) ? (_Float16)src[(size_t)r * C + c] : (_Float16)0.0f;
    }
}

__global__ void k_zero16(_Float16* p, size_t nel) {
    for (size_t i = (size_t)blockIdx.x * blockDim.x + threadIdx.x; i < nel;
         i += (size_t)gridDim.x * blockDim.x)
        p[i] = (_Float16)0.0f;
}

// ---------------- K2: LayerNorm + QKV GEMM ----------------
// block = 16 tokens of one batch, 256 threads (8 waves)
__global__ void k_ln_qkv(const float* __restrict__ x, const float* ln_w,
                         const float* ln_b, const _Float16* __restrict__ w16,
                         _Float16* q16, _Float16* k16, _Float16* vT16) {
    int blk = blockIdx.x;
    int bn = blk / (TT / 16);
    int t0 = (blk % (TT / 16)) * 16;
    __shared__ __attribute__((aligned(32))) _Float16 a16[16 * 224];
    int tid = threadIdx.x;
    // --- LayerNorm: 16 threads per row ---
    int r = tid >> 4, s = tid & 15;
    const float* xr = x + ((size_t)bn * TT + t0 + r) * DIM;
    float sum = 0.0f;
    for (int c = s; c < DIM; c += 16) sum += xr[c];
#pragma unroll
    for (int mk = 1; mk < 16; mk <<= 1) sum += __shfl_xor(sum, mk, 32);
    float mu = sum * (1.0f / DIM);
    float var = 0.0f;
    for (int c = s; c < DIM; c += 16) { float d = xr[c] - mu; var += d * d; }
#pragma unroll
    for (int mk = 1; mk < 16; mk <<= 1) var += __shfl_xor(var, mk, 32);
    float rstd = rsqrtf(var * (1.0f / DIM) + 1e-5f);
    for (int c = s; c < 224; c += 16) {
        float hv = (c < DIM) ? (xr[c] - mu) * rstd * ln_w[c] + ln_b[c] : 0.0f;
        a16[r * 224 + c] = (_Float16)hv;
    }
    __syncthreads();
    // --- GEMM: 41 col tiles (648 -> 656), K = 224 (7 chunks of 32) ---
    int wave = tid >> 5, lane = tid & 31, g = lane >> 4, nn = lane & 15;
    for (int ct = wave; ct < 41; ct += 8) {
        // prefetch all B fragments for this tile; sched barrier keeps the
        // load clause ahead of the WMMA chain (partial waits instead of stalls)
        v16h b[7];
#pragma unroll
        for (int kc = 0; kc < 7; ++kc)
            b[kc] = load_bfrag(w16 + (size_t)(ct * 16) * 224 + kc * 32, 224);
        SCHED_SPLIT();
        v8f acc = {};
#pragma unroll
        for (int kc = 0; kc < 7; ++kc)
            acc = wmma_f16(load_afrag(a16 + kc * 32, 224), b[kc], acc);
        // scatter store (j fixed per lane; only m varies with i)
        int j = ct * 16 + nn;
        if (j < 3 * DIM) {
            int sec = j / DIM, jj = j % DIM, head = jj / HD, hd = jj % HD;
            size_t hb = (size_t)bn * NH + head;
            if (sec < 2) {
                _Float16* dst = (sec ? k16 : q16) + ((hb * TT + t0) * HDP + hd);
#pragma unroll
                for (int i = 0; i < 8; ++i)
                    dst[(size_t)(i + 8 * g) * HDP] = (_Float16)acc[i];
            } else {
                _Float16* dst = vT16 + ((hb * HDP3 + hd) * TT + t0);
#pragma unroll
                for (int i = 0; i < 8; ++i)
                    dst[i + 8 * g] = (_Float16)acc[i];
            }
        }
    }
}

// ---------------- K3: flash attention with relative biases ----------------
// block = 64 q rows of one (n,h); 128 threads = 4 independent waves x 16 rows
__global__ void k_attn(const _Float16* __restrict__ q16, const _Float16* __restrict__ k16,
                       const _Float16* __restrict__ vT16, const float* __restrict__ tbl_t,
                       const float* __restrict__ bias_h, float* __restrict__ attn_out) {
    int blk = blockIdx.x;
    int qt = blk % (TT / 64);
    int nh = blk / (TT / 64);
    int h = nh % NH, n = nh / NH;
    int wave = threadIdx.x >> 5, lane = threadIdx.x & 31;
    int g = lane >> 4, nn = lane & 15;
    int qrow0 = qt * 64 + wave * 16;
    const _Float16* qbase = q16 + (((size_t)n * NH + h) * TT + qrow0) * HDP;
    const _Float16* kbase = k16 + (((size_t)n * NH + h) * TT) * HDP;
    const _Float16* vbase = vT16 + (((size_t)n * NH + h) * HDP3) * TT;
    const float* bh = bias_h + h * VJ * VJ;
    __shared__ __attribute__((aligned(32))) _Float16 pbuf[4][16 * 32];
    _Float16* pb = &pbuf[wave][0];

    v16h qf0 = load_afrag(qbase, HDP);
    v16h qf1 = load_afrag(qbase + 32, HDP);
    v8f o0 = {}, o1 = {}, o2 = {};
    float rmax[8], rsum[8];
    const float* bhq[8];   // hop-bias row per q row
    const float* ttq[8];   // temporal table row at k-frame 0 per q row
#pragma unroll
    for (int i = 0; i < 8; ++i) {
        rmax[i] = -1e30f; rsum[i] = 0.0f;
        int qtok = qrow0 + i + 8 * g;
        bhq[i] = bh + (qtok % VJ) * VJ;
        ttq[i] = tbl_t + (qtok / VJ + FF - 1) * NH + h;
    }
    // incremental div/mod-by-25 state for key indices (key0 advances by 32)
    int k0d = 0, k0m = nn;                       // (key0 + nn)
    int k1d = (nn + 16 >= VJ) ? 1 : 0;           // (key0 + 16 + nn)
    int k1m = (nn + 16 >= VJ) ? nn + 16 - VJ : nn + 16;

    for (int kt = 0; kt < TT / 32; ++kt) {
        int key0 = kt * 32;
        // issue ALL memory for this step up front: 4 K-frags + 3 V-frags
        v16h kb0 = load_bfrag(kbase + (size_t)key0 * HDP, HDP);
        v16h kb1 = load_bfrag(kbase + (size_t)key0 * HDP + 32, HDP);
        v16h kb2 = load_bfrag(kbase + (size_t)(key0 + 16) * HDP, HDP);
        v16h kb3 = load_bfrag(kbase + (size_t)(key0 + 16) * HDP + 32, HDP);
        v16h vb0 = load_bfrag(vbase + key0, TT);
        v16h vb1 = load_bfrag(vbase + (size_t)16 * TT + key0, TT);
        v16h vb2 = load_bfrag(vbase + (size_t)32 * TT + key0, TT);
        SCHED_SPLIT();

        v8f s0 = {}, s1 = {};
        s0 = wmma_f16(qf0, kb0, s0);
        s0 = wmma_f16(qf1, kb1, s0);
        s1 = wmma_f16(qf0, kb2, s1);
        s1 = wmma_f16(qf1, kb3, s1);

        int k0off = k0d * NH, k1off = k1d * NH;
#pragma unroll
        for (int i = 0; i < 8; ++i) {
            float b0 = bhq[i][k0m] + *(ttq[i] - k0off);
            float b1 = bhq[i][k1m] + *(ttq[i] - k1off);
            s0[i] = (s0[i] + b0) * (1.0f / 6.0f);
            s1[i] = (s1[i] + b1) * (1.0f / 6.0f);
        }
        // online softmax per row (16-lane groups hold one row)
        float corr[8];
#pragma unroll
        for (int i = 0; i < 8; ++i) {
            float mx = fmaxf(s0[i], s1[i]);
#pragma unroll
            for (int mk = 1; mk < 16; mk <<= 1) mx = fmaxf(mx, __shfl_xor(mx, mk, 32));
            float newm = fmaxf(rmax[i], mx);
            float p0 = __expf(s0[i] - newm), p1 = __expf(s1[i] - newm);
            float ts = p0 + p1;
#pragma unroll
            for (int mk = 1; mk < 16; mk <<= 1) ts += __shfl_xor(ts, mk, 32);
            float c = __expf(rmax[i] - newm);
            rsum[i] = rsum[i] * c + ts;
            rmax[i] = newm;
            corr[i] = c;
            s0[i] = p0; s1[i] = p1;
        }
        // stage P (16x32 f16) in per-wave LDS, reload as A-fragment
#pragma unroll
        for (int i = 0; i < 8; ++i) {
            int m = i + 8 * g;
            pb[m * 32 + nn] = (_Float16)s0[i];
            pb[m * 32 + 16 + nn] = (_Float16)s1[i];
        }
        asm volatile("s_wait_dscnt 0" ::: "memory");
        v16h pf = load_afrag(pb, 32);
#pragma unroll
        for (int i = 0; i < 8; ++i) { o0[i] *= corr[i]; o1[i] *= corr[i]; o2[i] *= corr[i]; }
        o0 = wmma_f16(pf, vb0, o0);
        o1 = wmma_f16(pf, vb1, o1);
        o2 = wmma_f16(pf, vb2, o2);
        // advance (key + 32) div/mod 25
        k0m += 7; if (k0m >= VJ) { k0m -= VJ; k0d += 2; } else { k0d += 1; }
        k1m += 7; if (k1m >= VJ) { k1m -= VJ; k1d += 2; } else { k1d += 1; }
    }
#pragma unroll
    for (int i = 0; i < 8; ++i) {
        float inv = 1.0f / rsum[i];
        int qtok = qrow0 + i + 8 * g;
        float* orow = attn_out + ((size_t)n * TT + qtok) * DIM + h * HD;
        orow[nn] = o0[i] * inv;
        orow[16 + nn] = o1[i] * inv;
        if (nn < 4) orow[32 + nn] = o2[i] * inv;
    }
}

// ---------------- K4: proj GEMM + bias + residual + LayerNorm ----------------
__global__ void k_proj(const float* __restrict__ x, const float* __restrict__ attn_out,
                       const _Float16* __restrict__ pw16, const float* pbias,
                       const float* n1w, const float* n1b, float* __restrict__ x1,
                       _Float16* __restrict__ h1_16) {
    int blk = blockIdx.x;
    int bn = blk / (TT / 16);
    int t0 = (blk % (TT / 16)) * 16;
    __shared__ __attribute__((aligned(32))) _Float16 a16[16 * 224];
    __shared__ float xs[16 * DIM];
    int tid = threadIdx.x;
    const float* ain = attn_out + ((size_t)bn * TT + t0) * DIM;
    for (int idx = tid; idx < 16 * 224; idx += 256) {
        int rr = idx / 224, cc = idx % 224;
        a16[idx] = (cc < DIM) ? (_Float16)ain[rr * DIM + cc] : (_Float16)0.0f;
    }
    __syncthreads();
    int wave = tid >> 5, lane = tid & 31, g = lane >> 4, nn = lane & 15;
    for (int ct = wave; ct < 14; ct += 8) {
        v16h b[7];
#pragma unroll
        for (int kc = 0; kc < 7; ++kc)
            b[kc] = load_bfrag(pw16 + (size_t)(ct * 16) * 224 + kc * 32, 224);
        SCHED_SPLIT();
        v8f acc = {};
#pragma unroll
        for (int kc = 0; kc < 7; ++kc)
            acc = wmma_f16(load_afrag(a16 + kc * 32, 224), b[kc], acc);
        int j = ct * 16 + nn;
        if (j < DIM) {
#pragma unroll
            for (int i = 0; i < 8; ++i) {
                int m = i + 8 * g;
                size_t off = ((size_t)bn * TT + t0 + m) * DIM + j;
                float v = acc[i] + pbias[j] + x[off];
                x1[off] = v;
                xs[m * DIM + j] = v;
            }
        }
    }
    __syncthreads();
    // LayerNorm on x1 tile -> h1 (f16, padded to 224)
    int r = tid >> 4, s = tid & 15;
    const float* xr = xs + r * DIM;
    float sum = 0.0f;
    for (int c = s; c < DIM; c += 16) sum += xr[c];
#pragma unroll
    for (int mk = 1; mk < 16; mk <<= 1) sum += __shfl_xor(sum, mk, 32);
    float mu = sum * (1.0f / DIM);
    float var = 0.0f;
    for (int c = s; c < DIM; c += 16) { float d = xr[c] - mu; var += d * d; }
#pragma unroll
    for (int mk = 1; mk < 16; mk <<= 1) var += __shfl_xor(var, mk, 32);
    float rstd = rsqrtf(var * (1.0f / DIM) + 1e-5f);
    _Float16* hrow = h1_16 + ((size_t)bn * TT + t0 + r) * 224;
    for (int c = s; c < 224; c += 16)
        hrow[c] = (c < DIM) ? (_Float16)((xr[c] - mu) * rstd * n1w[c] + n1b[c])
                            : (_Float16)0.0f;
}

// ---------------- K5: fc1 GEMM + bias + exact GELU ----------------
__global__ void k_fc1(const _Float16* __restrict__ h1_16, const _Float16* __restrict__ w16,
                      const float* b1, _Float16* __restrict__ m1_16) {
    size_t row0 = (size_t)blockIdx.x * 16;
    const _Float16* abase = h1_16 + row0 * 224;
    int tid = threadIdx.x, wave = tid >> 5, lane = tid & 31, g = lane >> 4, nn = lane & 15;
    // A fragments are identical for every column tile: load once
    v16h a[7];
#pragma unroll
    for (int kc = 0; kc < 7; ++kc) a[kc] = load_afrag(abase + kc * 32, 224);
    for (int ct = wave; ct < 54; ct += 8) {
        v16h b[7];
#pragma unroll
        for (int kc = 0; kc < 7; ++kc)
            b[kc] = load_bfrag(w16 + (size_t)(ct * 16) * 224 + kc * 32, 224);
        SCHED_SPLIT();
        v8f acc = {};
#pragma unroll
        for (int kc = 0; kc < 7; ++kc) acc = wmma_f16(a[kc], b[kc], acc);
        int j = ct * 16 + nn;
#pragma unroll
        for (int i = 0; i < 8; ++i) {
            int m = i + 8 * g;
            float v = acc[i] + b1[j];
            v = 0.5f * v * (1.0f + erff(v * 0.70710678118654752f));
            m1_16[(row0 + m) * (4 * DIM) + j] = (_Float16)v;
        }
    }
}

// ---------------- K6: fc2 GEMM + bias + residual ----------------
__global__ void k_fc2(const _Float16* __restrict__ m1_16, const _Float16* __restrict__ w16,
                      const float* b2, const float* __restrict__ x1, float* __restrict__ out) {
    size_t row0 = (size_t)blockIdx.x * 16;
    int tid = threadIdx.x, wave = tid >> 5, lane = tid & 31, g = lane >> 4, nn = lane & 15;
    // stage the 16x864 f16 A tile in LDS once (27 K-chunks is too many registers)
    __shared__ __attribute__((aligned(32))) _Float16 a16[16 * 4 * DIM];
    {
        const uint4* src = (const uint4*)(m1_16 + row0 * (4 * DIM));
        uint4* dst = (uint4*)a16;
        for (int idx = tid; idx < 16 * 4 * DIM / 8; idx += 256) dst[idx] = src[idx];
    }
    __syncthreads();
    for (int ct = wave; ct < 14; ct += 8) {
        const _Float16* wb = w16 + (size_t)(ct * 16) * (4 * DIM);
        v8f acc = {};
        // 27 K-chunks in 3 groups of 9: prefetch group loads, then WMMA chain
#pragma unroll
        for (int gb = 0; gb < 3; ++gb) {
            v16h b[9];
#pragma unroll
            for (int u = 0; u < 9; ++u)
                b[u] = load_bfrag(wb + (gb * 9 + u) * 32, 4 * DIM);
            SCHED_SPLIT();
#pragma unroll
            for (int u = 0; u < 9; ++u)
                acc = wmma_f16(load_afrag(a16 + (gb * 9 + u) * 32, 4 * DIM), b[u], acc);
        }
        int j = ct * 16 + nn;
        if (j < DIM) {
#pragma unroll
            for (int i = 0; i < 8; ++i) {
                int m = i + 8 * g;
                size_t off = (row0 + m) * DIM + j;
                out[off] = acc[i] + b2[j] + x1[off];
            }
        }
    }
}

// ---------------- workspace layout (bytes, 256-aligned) ----------------
#define O_TBLT   ((size_t)0)
#define O_BIASH  ((size_t)3072)
#define O_QKVW   ((size_t)18432)
#define O_PROJW  ((size_t)312320)
#define O_FC1W   ((size_t)412672)
#define O_FC2W   ((size_t)799744)
#define O_Q16    ((size_t)1186816)
#define O_K16    ((size_t)11017216)
#define O_VT16   ((size_t)20847616)
#define O_ATTN   ((size_t)28220416)
#define O_X1     ((size_t)39279616)
#define O_H1     ((size_t)50338816)
#define O_M1     ((size_t)56073216)

extern "C" void kernel_launch(void* const* d_in, const int* in_sizes, int n_in,
                              void* d_out, int out_size, void* d_ws, size_t ws_size,
                              hipStream_t stream) {
    (void)in_sizes; (void)n_in; (void)out_size; (void)ws_size;
    const float* x      = (const float*)d_in[0];
    const float* qkv_w  = (const float*)d_in[1];
    const float* proj_w = (const float*)d_in[2];
    const float* proj_b = (const float*)d_in[3];
    const float* norm_w = (const float*)d_in[4];
    const float* norm_b = (const float*)d_in[5];
    const float* n1_w   = (const float*)d_in[6];
    const float* n1_b   = (const float*)d_in[7];
    const float* fc1_w  = (const float*)d_in[8];
    const float* fc1_b  = (const float*)d_in[9];
    const float* fc2_w  = (const float*)d_in[10];
    const float* fc2_b  = (const float*)d_in[11];
    const float* t_w1   = (const float*)d_in[12];
    const float* t_b1   = (const float*)d_in[13];
    const float* t_w2   = (const float*)d_in[14];
    const float* t_b2   = (const float*)d_in[15];
    const float* h_w1   = (const float*)d_in[16];
    const float* h_b1   = (const float*)d_in[17];
    const float* h_w2   = (const float*)d_in[18];
    const float* h_b2   = (const float*)d_in[19];
    const int*   hop    = (const int*)d_in[20];

    char* ws = (char*)d_ws;
    float*     tbl_t  = (float*)(ws + O_TBLT);
    float*     bias_h = (float*)(ws + O_BIASH);
    _Float16*  qkvw16 = (_Float16*)(ws + O_QKVW);
    _Float16*  projw16= (_Float16*)(ws + O_PROJW);
    _Float16*  fc1w16 = (_Float16*)(ws + O_FC1W);
    _Float16*  fc2w16 = (_Float16*)(ws + O_FC2W);
    _Float16*  q16    = (_Float16*)(ws + O_Q16);
    _Float16*  k16    = (_Float16*)(ws + O_K16);
    _Float16*  vT16   = (_Float16*)(ws + O_VT16);
    float*     attn   = (float*)(ws + O_ATTN);
    float*     x1     = (float*)(ws + O_X1);
    _Float16*  h1_16  = (_Float16*)(ws + O_H1);
    _Float16*  m1_16  = (_Float16*)(ws + O_M1);
    float*     out    = (float*)d_out;

    k_tables<<<1, 256, 0, stream>>>(t_w1, t_b1, t_w2, t_b2, h_w1, h_b1, h_w2, h_b2,
                                    hop, tbl_t, bias_h);
    k_convpad<<<288, 256, 0, stream>>>(qkv_w, qkvw16, 648, 216, 656, 224);
    k_convpad<<<96, 256, 0, stream>>>(proj_w, projw16, 216, 216, 224, 224);
    k_convpad<<<288, 256, 0, stream>>>(fc1_w, fc1w16, 864, 216, 864, 224);
    k_convpad<<<288, 256, 0, stream>>>(fc2_w, fc2w16, 216, 864, 224, 864);
    // zero q16+k16 (contiguous) so hd pads 36..63 are zero
    k_zero16<<<2048, 256, 0, stream>>>(q16, (size_t)2 * 8 * NH * TT * HDP);

    k_ln_qkv<<<8 * (TT / 16), 256, 0, stream>>>(x, norm_w, norm_b, qkvw16, q16, k16, vT16);
    k_attn<<<8 * NH * (TT / 64), 128, 0, stream>>>(q16, k16, vT16, tbl_t, bias_h, attn);
    k_proj<<<8 * (TT / 16), 256, 0, stream>>>(x, attn, projw16, proj_b, n1_w, n1_b, x1, h1_16);
    k_fc1<<<8 * (TT / 16), 256, 0, stream>>>(h1_16, fc1w16, fc1_b, m1_16);
    k_fc2<<<8 * (TT / 16), 256, 0, stream>>>(m1_16, fc2w16, fc2_b, x1, out);
}
